// QuantizedLinear_12481174962914
// MI455X (gfx1250) — matrix operand order repaired and checked by reading, not compiled, not run on hardware
//
#include <hip/hip_runtime.h>

typedef __attribute__((ext_vector_type(16))) __bf16       v16bf;
typedef __attribute__((ext_vector_type(8)))  float        v8f;
typedef __attribute__((ext_vector_type(4)))  unsigned int u32x4;
typedef __attribute__((ext_vector_type(4)))  float        f32x4;
typedef __attribute__((ext_vector_type(4)))  int          i32x4;

#define IN_DIM   4096
#define OUT_DIM  4096
#define BK       32
#define NK       (IN_DIM / BK)       /* 128 K-steps */
#define LDSROW   48                  /* bf16 elems per LDS row: 32 data + 16 pad */
#define TILEE    (128 * LDSROW)      /* elements per LDS tile buffer */
#define TILEB    (TILEE * 2)         /* bytes per LDS tile buffer (12288) */

__device__ __constant__ float NF4_CODE_D[16] = {
    -1.0f, -0.6961928009986877f, -0.5250730514526367f, -0.39491748809814453f,
    -0.28444138169288635f, -0.18477343022823334f, -0.09105003625154495f, 0.0f,
    0.07958029955625534f, 0.16093020141124725f, 0.24611230194568634f,
    0.33791524171829224f, 0.44070982933044434f, 0.5626170039176941f,
    0.7229568362236023f, 1.0f };

// round-to-nearest-even f32 -> bf16, two at a time, packed into one dword
__device__ __forceinline__ unsigned int pack2_bf16(float lo, float hi) {
    unsigned int ul = __builtin_bit_cast(unsigned int, lo);
    unsigned int uh = __builtin_bit_cast(unsigned int, hi);
    ul += 0x7FFFu + ((ul >> 16) & 1u);
    uh += 0x7FFFu + ((uh >> 16) & 1u);
    return (ul >> 16) | (uh & 0xFFFF0000u);
}

union FragBF16 { u32x4 q[2]; v16bf v; };

// ---- CDNA5 async-copy helpers (ASYNCcnt-tracked DMA: global -> LDS) --------
__device__ __forceinline__ void async_g2l_b128(unsigned lds_off,
                                               const void* gaddr) {
    asm volatile("global_load_async_to_lds_b128 %0, %1, off"
                 :: "v"(lds_off), "v"((unsigned long long)(size_t)gaddr)
                 : "memory");
}
__device__ __forceinline__ void wait_async0() {
    asm volatile("s_wait_asynccnt 0x0" ::: "memory");
}

// ============================================================================
// Pass 1a: NF4 dequant  W(int32 nibble + absmax) -> bf16   (memory bound)
// ============================================================================
__global__ __launch_bounds__(256)
void nf4_dequant_w(const int* __restrict__ qw, const float* __restrict__ absmax,
                   unsigned short* __restrict__ wb)
{
    __shared__ float lut[16];
    const int t = threadIdx.x;
    if (t < 16) lut[t] = NF4_CODE_D[t];
    __syncthreads();

    const size_t base = ((size_t)blockIdx.x * 256 + t) * 8;  // 8 elems/thread
    i32x4 q0 = *(const i32x4*)(qw + base);
    i32x4 q1 = *(const i32x4*)(qw + base + 4);
    const float am = absmax[base >> 6];   // aligned 8-seg inside one 64-block

    u32x4 v;
    v.x = pack2_bf16(lut[q0.x & 15] * am, lut[q0.y & 15] * am);
    v.y = pack2_bf16(lut[q0.z & 15] * am, lut[q0.w & 15] * am);
    v.z = pack2_bf16(lut[q1.x & 15] * am, lut[q1.y & 15] * am);
    v.w = pack2_bf16(lut[q1.z & 15] * am, lut[q1.w & 15] * am);
    *(u32x4*)(wb + base) = v;
}

// ============================================================================
// Pass 1b: X f32 -> bf16   (memory bound)
// ============================================================================
__global__ __launch_bounds__(256)
void cvt_x_bf16(const float* __restrict__ x, unsigned short* __restrict__ xb)
{
    const size_t base = ((size_t)blockIdx.x * 256 + threadIdx.x) * 8;
    f32x4 a0 = *(const f32x4*)(x + base);
    f32x4 a1 = *(const f32x4*)(x + base + 4);
    u32x4 v;
    v.x = pack2_bf16(a0.x, a0.y);
    v.y = pack2_bf16(a0.z, a0.w);
    v.z = pack2_bf16(a1.x, a1.y);
    v.w = pack2_bf16(a1.z, a1.w);
    *(u32x4*)(xb + base) = v;
}

// ============================================================================
// Pass 2: bf16 GEMM, async-DMA staging, WMMA compute.
//   block = 256 thr = 8 waves, 128x128 block tile, wave = 64x32 (4x2 WMMA)
// ============================================================================
__global__ __launch_bounds__(256, 2)
void gemm_bf16_wmma_async(const unsigned short* __restrict__ xb,
                          const unsigned short* __restrict__ wb,
                          float* __restrict__ out)
{
    __shared__ __align__(16) unsigned short sA[2 * TILEE];
    __shared__ __align__(16) unsigned short sB[2 * TILEE];

    const int t    = threadIdx.x;
    const int bn   = blockIdx.x;   // N tile (fastest -> W L2-resident)
    const int bm   = blockIdx.y;   // M tile
    const int wave = t >> 5;
    const int lane = t & 31;
    const int wm   = wave >> 2;    // 0..1
    const int wn   = wave & 3;     // 0..3
    const int r    = lane & 15;
    const int hi   = lane >> 4;

    // async staging: thread t copies 2x16B chunks per operand per K-step
    const int tr = t >> 1;         // LDS row 0..127
    const int j0 = (t & 1) * 2;    // 16B-chunk index (0 or 2) of 4 per row
    const unsigned short* xp = xb + (size_t)(bm * 128 + tr) * IN_DIM + j0 * 8;
    const unsigned short* wp = wb + (size_t)(bn * 128 + tr) * IN_DIM + j0 * 8;
    const unsigned ldsA = (unsigned)(size_t)&sA[0] + tr * (LDSROW * 2) + j0 * 16;
    const unsigned ldsB = (unsigned)(size_t)&sB[0] + tr * (LDSROW * 2) + j0 * 16;

    v8f acc[4][2];
    #pragma unroll
    for (int mt = 0; mt < 4; ++mt)
        #pragma unroll
        for (int nt = 0; nt < 2; ++nt) {
            v8f z = {0.f,0.f,0.f,0.f,0.f,0.f,0.f,0.f};
            acc[mt][nt] = z;
        }

    // fragment-load bases (element offsets into flattened sA/sB, sans buffer)
    const int aRow = (wm * 64 + r) * LDSROW;
    const int bRow = (wn * 32 + r) * LDSROW + hi * 8;

    auto compute = [&](int bufe) {  // bufe = buffer element offset (0 / TILEE)
        FragBF16 a[4], b[2];
        #pragma unroll
        for (int mt = 0; mt < 4; ++mt) {
            const unsigned short* p = &sA[bufe + aRow + mt * (16 * LDSROW)];
            a[mt].q[0] = *(const u32x4*)(p + hi * 8);
            a[mt].q[1] = *(const u32x4*)(p + 16 + hi * 8);
        }
        #pragma unroll
        for (int nt = 0; nt < 2; ++nt) {
            const unsigned short* p = &sB[bufe + bRow + nt * (16 * LDSROW)];
            b[nt].q[0] = *(const u32x4*)(p);
            b[nt].q[1] = *(const u32x4*)(p + 16);
        }
        #pragma unroll
        for (int mt = 0; mt < 4; ++mt)
            #pragma unroll
            for (int nt = 0; nt < 2; ++nt)
                acc[mt][nt] = __builtin_amdgcn_wmma_f32_16x16x32_bf16(
                    false, a[mt].v, false, b[nt].v,
                    (short)0, acc[mt][nt], false, false);
    };

    // prologue DMA: tile 0 -> buffer 0
    async_g2l_b128(ldsA,      xp);
    async_g2l_b128(ldsA + 16, xp + 8);
    async_g2l_b128(ldsB,      wp);
    async_g2l_b128(ldsB + 16, wp + 8);

    unsigned bufB = 0;             // byte offset of buffer being computed
    int      bufE = 0;             // element offset of same

    #pragma unroll 1
    for (int kt = 0; kt < NK - 1; ++kt) {
        xp += BK;                  // next K tile
        wp += BK;
        wait_async0();             // own copies of tile kt landed in LDS
        __syncthreads();           // all copies landed; kt-1 compute done
        const unsigned nb = bufB ^ TILEB;
        async_g2l_b128(ldsA + nb,      xp);      // DMA kt+1 overlaps WMMAs
        async_g2l_b128(ldsA + nb + 16, xp + 8);
        async_g2l_b128(ldsB + nb,      wp);
        async_g2l_b128(ldsB + nb + 16, wp + 8);
        compute(bufE);
        bufB = nb;
        bufE ^= TILEE;
    }
    wait_async0();                 // final tile
    __syncthreads();
    compute(bufE);

    // C/D layout: VGPR e, lane-half hi -> m = e + 8*hi ; n = r
    float* obase = out + (size_t)(bm * 128 + wm * 64) * OUT_DIM + bn * 128 + wn * 32;
    #pragma unroll
    for (int mt = 0; mt < 4; ++mt)
        #pragma unroll
        for (int nt = 0; nt < 2; ++nt)
            #pragma unroll
            for (int e = 0; e < 8; ++e)
                obase[(size_t)(mt * 16 + hi * 8 + e) * OUT_DIM + nt * 16 + r] =
                    acc[mt][nt][e];
}

// ============================================================================
// Fallback: fused dequant+convert+GEMM (no scratch dependence)
// ============================================================================
__global__ __launch_bounds__(256, 2)
void nf4_linear_wmma_fused(const float* __restrict__ x,
                           const int*   __restrict__ qw,
                           const float* __restrict__ absmax,
                           float*       __restrict__ out)
{
    __shared__ __align__(16) unsigned short sA[2 * TILEE];
    __shared__ __align__(16) unsigned short sB[2 * TILEE];
    __shared__ float lut[16];

    const int t    = threadIdx.x;
    const int bn   = blockIdx.x;
    const int bm   = blockIdx.y;
    const int wave = t >> 5;
    const int lane = t & 31;
    const int wm   = wave >> 2;
    const int wn   = wave & 3;
    const int r    = lane & 15;
    const int hi   = lane >> 4;

    if (t < 16) lut[t] = NF4_CODE_D[t];

    const int tr  = t >> 1;
    const int seg = t & 1;
    const float* xrow = x  + (size_t)(bm * 128 + tr) * IN_DIM + seg * 16;
    const int*   qrow = qw + (size_t)(bn * 128 + tr) * IN_DIM + seg * 16;
    const size_t amax_base = (size_t)(bn * 128 + tr) * (IN_DIM / 64);

    f32x4 axr[4];
    i32x4 qvr[4];
    float amr;

    auto stage_load = [&](int kt) {
        const float* xp = xrow + kt * BK;
        const int*   qp = qrow + kt * BK;
        #pragma unroll
        for (int i = 0; i < 4; ++i) axr[i] = *(const f32x4*)(xp + i * 4);
        #pragma unroll
        for (int i = 0; i < 4; ++i) qvr[i] = *(const i32x4*)(qp + i * 4);
        amr = absmax[amax_base + ((unsigned)(kt * BK + seg * 16) >> 6)];
    };

    auto stage_store = [&](int bufe) {
        unsigned int pk[8];
        #pragma unroll
        for (int i = 0; i < 4; ++i) {
            pk[2*i]   = pack2_bf16(axr[i].x, axr[i].y);
            pk[2*i+1] = pack2_bf16(axr[i].z, axr[i].w);
        }
        u32x4* pa = (u32x4*)&sA[bufe + tr * LDSROW + seg * 16];
        { u32x4 v0 = {pk[0],pk[1],pk[2],pk[3]}; u32x4 v1 = {pk[4],pk[5],pk[6],pk[7]};
          pa[0] = v0; pa[1] = v1; }
        #pragma unroll
        for (int i = 0; i < 4; ++i) {
            float w0 = lut[qvr[i].x & 15] * amr;
            float w1 = lut[qvr[i].y & 15] * amr;
            float w2 = lut[qvr[i].z & 15] * amr;
            float w3 = lut[qvr[i].w & 15] * amr;
            pk[2*i]   = pack2_bf16(w0, w1);
            pk[2*i+1] = pack2_bf16(w2, w3);
        }
        u32x4* pb = (u32x4*)&sB[bufe + tr * LDSROW + seg * 16];
        { u32x4 v0 = {pk[0],pk[1],pk[2],pk[3]}; u32x4 v1 = {pk[4],pk[5],pk[6],pk[7]};
          pb[0] = v0; pb[1] = v1; }
    };

    v8f acc[4][2];
    #pragma unroll
    for (int mt = 0; mt < 4; ++mt)
        #pragma unroll
        for (int nt = 0; nt < 2; ++nt) {
            v8f z = {0.f,0.f,0.f,0.f,0.f,0.f,0.f,0.f};
            acc[mt][nt] = z;
        }

    const int aRow = (wm * 64 + r) * LDSROW;
    const int bRow = (wn * 32 + r) * LDSROW + hi * 8;

    auto compute = [&](int bufe) {
        FragBF16 a[4], b[2];
        #pragma unroll
        for (int mt = 0; mt < 4; ++mt) {
            const unsigned short* p = &sA[bufe + aRow + mt * (16 * LDSROW)];
            a[mt].q[0] = *(const u32x4*)(p + hi * 8);
            a[mt].q[1] = *(const u32x4*)(p + 16 + hi * 8);
        }
        #pragma unroll
        for (int nt = 0; nt < 2; ++nt) {
            const unsigned short* p = &sB[bufe + bRow + nt * (16 * LDSROW)];
            b[nt].q[0] = *(const u32x4*)(p);
            b[nt].q[1] = *(const u32x4*)(p + 16);
        }
        #pragma unroll
        for (int mt = 0; mt < 4; ++mt)
            #pragma unroll
            for (int nt = 0; nt < 2; ++nt)
                acc[mt][nt] = __builtin_amdgcn_wmma_f32_16x16x32_bf16(
                    false, a[mt].v, false, b[nt].v,
                    (short)0, acc[mt][nt], false, false);
    };

    stage_load(0);
    __syncthreads();
    stage_store(0);

    int bufE = 0;
    #pragma unroll 1
    for (int kt = 0; kt < NK - 1; ++kt) {
        __syncthreads();
        stage_load(kt + 1);
        compute(bufE);
        stage_store(bufE ^ TILEE);
        bufE ^= TILEE;
    }
    __syncthreads();
    compute(bufE);

    float* obase = out + (size_t)(bm * 128 + wm * 64) * OUT_DIM + bn * 128 + wn * 32;
    #pragma unroll
    for (int mt = 0; mt < 4; ++mt)
        #pragma unroll
        for (int nt = 0; nt < 2; ++nt)
            #pragma unroll
            for (int e = 0; e < 8; ++e)
                obase[(size_t)(mt * 16 + hi * 8 + e) * OUT_DIM + nt * 16 + r] =
                    acc[mt][nt][e];
}

extern "C" void kernel_launch(void* const* d_in, const int* in_sizes, int n_in,
                              void* d_out, int out_size, void* d_ws, size_t ws_size,
                              hipStream_t stream) {
    const float* x      = (const float*)d_in[0];
    const int*   qw     = (const int*)d_in[1];
    const float* absmax = (const float*)d_in[2];
    float*       out    = (float*)d_out;

    const int M = in_sizes[0] / IN_DIM;            // 16384
    const size_t needW = (size_t)OUT_DIM * IN_DIM * 2;     // 33.5 MB
    const size_t needX = (size_t)M * IN_DIM * 2;           // 134 MB
    dim3 grid(OUT_DIM / 128, M / 128);             // (32, 128)

    if (d_ws != nullptr && ws_size >= needW + needX) {
        unsigned short* wbuf = (unsigned short*)d_ws;
        unsigned short* xbuf = (unsigned short*)((char*)d_ws + needW);

        nf4_dequant_w<<<(OUT_DIM * (size_t)IN_DIM) / 2048, 256, 0, stream>>>(
            qw, absmax, wbuf);
        cvt_x_bf16<<<((size_t)M * IN_DIM) / 2048, 256, 0, stream>>>(x, xbuf);
        gemm_bf16_wmma_async<<<grid, dim3(256), 0, stream>>>(xbuf, wbuf, out);
    } else {
        nf4_linear_wmma_fused<<<grid, dim3(256), 0, stream>>>(x, qw, absmax, out);
    }
}
